// GRUModel_40183714022112
// MI455X (gfx1250) — matrix operand order repaired and checked by reading, not compiled, not run on hardware
//
#include <hip/hip_runtime.h>
#include <hip/hip_bf16.h>

// ---------------------------------------------------------------------------
// GRU LM forward for MI455X (gfx1250, wave32, WMMA bf16 path)
// V=32000, E=256, H=256, O=32000, B=256, S=512
// ---------------------------------------------------------------------------

#define V_ 32000
#define E_ 256
#define H_ 256
#define O_ 32000
#define B_ 256
#define S_ 512

typedef __attribute__((ext_vector_type(16))) __bf16 bf16x16;
typedef __attribute__((ext_vector_type(8)))  float  f32x8;

static __device__ __forceinline__ f32x8 wmma_bf16(bf16x16 a, bf16x16 b, f32x8 c) {
  // D = A(16x32 bf16) * B(32x16 bf16) + C(16x16 f32)
  return __builtin_amdgcn_wmma_f32_16x16x32_bf16(
      /*neg_a=*/false, a, /*neg_b=*/false, b,
      /*c_mod=*/(short)0, c, /*reuse_a=*/false, /*reuse_b=*/false);
}

// A-fragment (16x32, 16-bit): lane L owns row M = L%16.
// Lanes 0-15 hold K = {k0+0..7, k0+16..23}; lanes 16-31 hold K = {k0+8..15, k0+24..31}.
static __device__ __forceinline__ bf16x16 load_a_frag_f32row(const float* rowptr,
                                                             int lane, int k0) {
  int half = (lane >> 4) & 1;
  const float* p0 = rowptr + k0 + half * 8;
  const float* p1 = p0 + 16;
  bf16x16 a;
#pragma unroll
  for (int i = 0; i < 8; ++i) {
    a[i]     = (__bf16)p0[i];
    a[8 + i] = (__bf16)p1[i];
  }
  return a;
}

static __device__ __forceinline__ bf16x16 load_a_frag_bf16(const __bf16* A, int ld,
                                                           int lane, int k0) {
  int row  = lane & 15;
  int half = (lane >> 4) & 1;
  const __bf16* p0 = A + row * ld + k0 + half * 8;
  const __bf16* p1 = p0 + 16;
  bf16x16 a;
#pragma unroll
  for (int i = 0; i < 8; ++i) {
    a[i]     = p0[i];
    a[8 + i] = p1[i];
  }
  return a;
}

// B-fragment (32x16, 16-bit): lane L owns column N = n0 + L%16,
// K = {k0+(L/16)*8 .. +7, k0+16+(L/16)*8 .. +7} (same as A). W is row-major [N_total, K] so
// B[k][n] = W[n][k].
static __device__ __forceinline__ bf16x16 load_b_frag_bf16(const __bf16* W, int ldk,
                                                           int n0, int k0, int lane) {
  int n  = n0 + (lane & 15);
  int kb = k0 + ((lane >> 4) << 3);
  const __bf16* p = W + (long)n * ldk + kb;
  bf16x16 b;
#pragma unroll
  for (int i = 0; i < 8; ++i) { b[i] = p[i]; b[8 + i] = p[16 + i]; }
  return b;
}

// C/D layout: VGPR i: lanes 0-15 -> M=i, N=lane; lanes 16-31 -> M=8+i, N=lane-16.

// ---------------------------------------------------------------------------
// Prep kernels
// ---------------------------------------------------------------------------
__global__ void __launch_bounds__(256) cvt_f32_bf16_kernel(const float* __restrict__ src,
                                                           __bf16* __restrict__ dst, int n) {
  int i = blockIdx.x * 256 + threadIdx.x;
  if (i < n) dst[i] = (__bf16)src[i];
}

__global__ void __launch_bounds__(256) zero_f32_kernel(float* __restrict__ p, int n) {
  int i = blockIdx.x * 256 + threadIdx.x;
  if (i < n) p[i] = 0.0f;
}

// ---------------------------------------------------------------------------
// Kernel A: fused embedding gather + input projections.
// Each wave owns one 16-row M-tile of the flat token stream (M = B*S = 131072),
// preloads 8 A-fragments (full K=256) and loops over 48 N-tiles (3 gates x 16).
// Output proj stored bf16 as [3][S][B][H].
// ---------------------------------------------------------------------------
__global__ void __launch_bounds__(128) proj_kernel(const int* __restrict__ x,
                                                   const float* __restrict__ emb,
                                                   const __bf16* __restrict__ Wb, // [3][H][E]
                                                   const float* __restrict__ bz,
                                                   const float* __restrict__ br,
                                                   const float* __restrict__ bh,
                                                   __bf16* __restrict__ proj) {
  int wave  = threadIdx.x >> 5;
  int lane  = threadIdx.x & 31;
  int mtile = blockIdx.x * 4 + wave;   // 8192 tiles total
  int m0    = mtile * 16;

  int row = lane & 15;
  int tok = x[m0 + row];
  const float* rowptr = emb + (long)tok * E_;

  bf16x16 afrag[8];
#pragma unroll
  for (int kk = 0; kk < 8; ++kk) afrag[kk] = load_a_frag_f32row(rowptr, lane, kk * 32);

  const float* biases[3] = {bz, br, bh};
  int col   = lane & 15;
  int rbase = (lane >> 4) * 8;

  for (int g = 0; g < 3; ++g) {
    const __bf16* W = Wb + (long)g * H_ * E_;
    const float* bias_v = biases[g];
    for (int nt = 0; nt < 16; ++nt) {
      f32x8 acc = {};
#pragma unroll
      for (int kk = 0; kk < 8; ++kk) {
        bf16x16 bfrag = load_b_frag_bf16(W, E_, nt * 16, kk * 32, lane);
        acc = wmma_bf16(afrag[kk], bfrag, acc);
      }
      int n = nt * 16 + col;
      float bias = bias_v[n];
#pragma unroll
      for (int i = 0; i < 8; ++i) {
        int m = m0 + rbase + i;
        int b = m >> 9;       // / S_
        int s = m & (S_ - 1); // % S_
        proj[(((long)g * S_ + s) * B_ + b) * H_ + n] = (__bf16)(acc[i] + bias);
      }
    }
  }
}

// ---------------------------------------------------------------------------
// Kernel B: one GRU time step. 16 workgroups (one per 16-row batch tile),
// 8 waves each. Self-contained per tile: z,r via WMMA, r*h into LDS (bf16),
// barrier, h_tilde via WMMA, blend, store h_next.
// ---------------------------------------------------------------------------
__global__ void __launch_bounds__(256) gru_step_kernel(const float* __restrict__ hprev,
                                                       float* __restrict__ hnext,
                                                       const __bf16* __restrict__ Ub, // [3][H][H]
                                                       const float* __restrict__ buz,
                                                       const float* __restrict__ bur,
                                                       const float* __restrict__ buh,
                                                       const __bf16* __restrict__ proj,
                                                       int s) {
  __shared__ __bf16 h_bf[16 * H_];
  __shared__ __bf16 rh_bf[16 * H_];
  __shared__ float  z_sh[16 * H_];

  int b0  = blockIdx.x * 16;
  int tid = threadIdx.x;

  // stage h_prev tile into LDS as bf16
  for (int idx = tid; idx < 16 * H_; idx += 256) {
    int rr = idx >> 8;
    int cc = idx & (H_ - 1);
    h_bf[idx] = (__bf16)hprev[(b0 + rr) * H_ + cc];
  }
  __syncthreads();

  int wave = tid >> 5, lane = tid & 31;
  const __bf16* Uz = Ub;
  const __bf16* Ur = Ub + H_ * H_;
  const __bf16* Uh = Ub + 2 * H_ * H_;
  const __bf16* pz = proj + ((long)0 * S_ + s) * B_ * H_;
  const __bf16* pr = proj + ((long)1 * S_ + s) * B_ * H_;
  const __bf16* ph = proj + ((long)2 * S_ + s) * B_ * H_;

  bf16x16 ha[8];
#pragma unroll
  for (int kk = 0; kk < 8; ++kk) ha[kk] = load_a_frag_bf16(h_bf, H_, lane, kk * 32);

  int col   = lane & 15;
  int rbase = (lane >> 4) * 8;

  // phase 1: z and r for this wave's two N-tiles
  for (int t = 0; t < 2; ++t) {
    int n0 = (wave + t * 8) * 16;
    f32x8 az = {}, ar = {};
#pragma unroll
    for (int kk = 0; kk < 8; ++kk) {
      bf16x16 bzf = load_b_frag_bf16(Uz, H_, n0, kk * 32, lane);
      az = wmma_bf16(ha[kk], bzf, az);
      bf16x16 brf = load_b_frag_bf16(Ur, H_, n0, kk * 32, lane);
      ar = wmma_bf16(ha[kk], brf, ar);
    }
    int n = n0 + col;
    float bzv = buz[n], brv = bur[n];
#pragma unroll
    for (int i = 0; i < 8; ++i) {
      int r_loc = rbase + i;
      int b     = b0 + r_loc;
      float iz  = (float)pz[(long)b * H_ + n];
      float ir  = (float)pr[(long)b * H_ + n];
      float zv  = 1.0f / (1.0f + __expf(-(az[i] + iz + bzv)));
      float rv  = 1.0f / (1.0f + __expf(-(ar[i] + ir + brv)));
      float hp  = hprev[(long)b * H_ + n];
      z_sh[r_loc * H_ + n]  = zv;
      rh_bf[r_loc * H_ + n] = (__bf16)(rv * hp);
    }
  }
  __syncthreads();

  // phase 2: h_tilde and blend
  bf16x16 rha[8];
#pragma unroll
  for (int kk = 0; kk < 8; ++kk) rha[kk] = load_a_frag_bf16(rh_bf, H_, lane, kk * 32);

  for (int t = 0; t < 2; ++t) {
    int n0 = (wave + t * 8) * 16;
    f32x8 ah = {};
#pragma unroll
    for (int kk = 0; kk < 8; ++kk) {
      bf16x16 bhf = load_b_frag_bf16(Uh, H_, n0, kk * 32, lane);
      ah = wmma_bf16(rha[kk], bhf, ah);
    }
    int n = n0 + col;
    float bhv = buh[n];
#pragma unroll
    for (int i = 0; i < 8; ++i) {
      int r_loc = rbase + i;
      int b     = b0 + r_loc;
      float ih  = (float)ph[(long)b * H_ + n];
      float ht  = tanhf(ah[i] + ih + bhv);
      float zv  = z_sh[r_loc * H_ + n];
      float hp  = hprev[(long)b * H_ + n];
      hnext[(long)b * H_ + n] = (1.0f - zv) * hp + zv * ht;
    }
  }
}

// ---------------------------------------------------------------------------
// Kernel C: logits = h_last @ Wf^T + bf.  M=256, N=32000, K=256.
// 8 waves per block, each wave one 16x16 tile.
// ---------------------------------------------------------------------------
__global__ void __launch_bounds__(256) final_kernel(const float* __restrict__ hlast,
                                                    const __bf16* __restrict__ Wfb, // [O,H]
                                                    const float* __restrict__ bfv,
                                                    float* __restrict__ out) {
  int wave  = threadIdx.x >> 5;
  int lane  = threadIdx.x & 31;
  int ntile = blockIdx.x * 8 + wave;   // gridDim.x = 250 -> 2000 N-tiles
  int mtile = blockIdx.y;              // 16 M-tiles
  int m0 = mtile * 16, n0 = ntile * 16;

  int row = lane & 15;
  const float* rowptr = hlast + (long)(m0 + row) * H_;

  f32x8 acc = {};
#pragma unroll
  for (int kk = 0; kk < 8; ++kk) {
    bf16x16 a = load_a_frag_f32row(rowptr, lane, kk * 32);
    bf16x16 b = load_b_frag_bf16(Wfb, H_, n0, kk * 32, lane);
    acc = wmma_bf16(a, b, acc);
  }
  int col   = lane & 15;
  int rbase = (lane >> 4) * 8;
  int n     = n0 + col;
  float bias = bfv[n];
#pragma unroll
  for (int i = 0; i < 8; ++i) {
    int m = m0 + rbase + i;
    out[(long)m * O_ + n] = acc[i] + bias;
  }
}

// ---------------------------------------------------------------------------
// Host-side launcher
// ---------------------------------------------------------------------------
extern "C" void kernel_launch(void* const* d_in, const int* in_sizes, int n_in,
                              void* d_out, int out_size, void* d_ws, size_t ws_size,
                              hipStream_t stream) {
  (void)in_sizes; (void)n_in; (void)out_size; (void)ws_size;

  const int*   x   = (const int*)d_in[0];
  const float* emb = (const float*)d_in[1];
  const float* Wz  = (const float*)d_in[2];
  const float* bz  = (const float*)d_in[3];
  const float* Uz  = (const float*)d_in[4];
  const float* buz = (const float*)d_in[5];
  const float* Wr  = (const float*)d_in[6];
  const float* br  = (const float*)d_in[7];
  const float* Ur  = (const float*)d_in[8];
  const float* bur = (const float*)d_in[9];
  const float* Wh  = (const float*)d_in[10];
  const float* bh  = (const float*)d_in[11];
  const float* Uh  = (const float*)d_in[12];
  const float* buh = (const float*)d_in[13];
  const float* Wf  = (const float*)d_in[14];
  const float* bfv = (const float*)d_in[15];
  float* out = (float*)d_out;

  // workspace carve-up
  char* ws = (char*)d_ws;
  __bf16* proj = (__bf16*)ws;                                  // [3][S][B][H] bf16
  size_t off = (size_t)3 * S_ * B_ * H_ * sizeof(__bf16);
  __bf16* Wb  = (__bf16*)(ws + off); off += (size_t)3 * H_ * E_ * sizeof(__bf16);
  __bf16* Ub  = (__bf16*)(ws + off); off += (size_t)3 * H_ * H_ * sizeof(__bf16);
  __bf16* Wfb = (__bf16*)(ws + off); off += (size_t)O_ * H_ * sizeof(__bf16);
  float*  h0  = (float*)(ws + off);  off += (size_t)B_ * H_ * sizeof(float);
  float*  h1  = (float*)(ws + off);

  const int HE = H_ * E_;   // 65536
  const int HH = H_ * H_;   // 65536
  dim3 cvtb(256);
  // input-side weights -> bf16
  cvt_f32_bf16_kernel<<<(HE + 255) / 256, cvtb, 0, stream>>>(Wz, Wb, HE);
  cvt_f32_bf16_kernel<<<(HE + 255) / 256, cvtb, 0, stream>>>(Wr, Wb + HE, HE);
  cvt_f32_bf16_kernel<<<(HE + 255) / 256, cvtb, 0, stream>>>(Wh, Wb + 2 * HE, HE);
  // recurrent weights -> bf16
  cvt_f32_bf16_kernel<<<(HH + 255) / 256, cvtb, 0, stream>>>(Uz, Ub, HH);
  cvt_f32_bf16_kernel<<<(HH + 255) / 256, cvtb, 0, stream>>>(Ur, Ub + HH, HH);
  cvt_f32_bf16_kernel<<<(HH + 255) / 256, cvtb, 0, stream>>>(Uh, Ub + 2 * HH, HH);
  // output weights -> bf16
  const int OF = O_ * H_;   // 8192000
  cvt_f32_bf16_kernel<<<(OF + 255) / 256, cvtb, 0, stream>>>(Wf, Wfb, OF);
  // h0 = 0 (workspace is poisoned; must re-init every call)
  zero_f32_kernel<<<(B_ * H_ + 255) / 256, cvtb, 0, stream>>>(h0, B_ * H_);

  // fused gather + input projections: 131072/16 = 8192 M-tiles, 4 waves/block
  proj_kernel<<<8192 / 4, 128, 0, stream>>>(x, emb, Wb, bz, br, bh, proj);

  // sequential recurrence: one launch per step, ping-pong h buffers
  for (int s = 0; s < S_; ++s) {
    const float* hp = (s & 1) ? h1 : h0;
    float*       hn = (s & 1) ? h0 : h1;
    gru_step_kernel<<<B_ / 16, 256, 0, stream>>>(hp, hn, Ub, buz, bur, buh, proj, s);
  }
  // S_=512 is even -> final state lands in h0

  final_kernel<<<dim3(O_ / 16 / 8, B_ / 16), 256, 0, stream>>>(h0, Wfb, bfv, out);
}
